// FusedEPMoE_60601988547245
// MI455X (gfx1250) — matrix operand order repaired and checked by reading, not compiled, not run on hardware
//
#include <hip/hip_runtime.h>

// ---------------- problem constants ----------------
constexpr int cT = 4096;   // tokens
constexpr int cH = 2048;   // hidden
constexpr int cF = 1408;   // intermediate
constexpr int cE = 8;      // experts
constexpr int cK = 2;      // top_k

// ---------------- workspace layout (bytes) ----------------
constexpr size_t OFF_CNT  = 0;                                  // int[8]
constexpr size_t OFF_BASE = 256;                                // int[9]
constexpr size_t OFF_CUR  = 512;                                // int[8]
constexpr size_t OFF_EID  = 1024;                               // int[T*K]
constexpr size_t OFF_TW   = OFF_EID + (size_t)cT * cK * 4;      // float[T*K]
constexpr size_t OFF_STK  = OFF_TW  + (size_t)cT * cK * 4;      // int[T*K]  (token*2+k per slot)
constexpr size_t OFF_SW   = OFF_STK + (size_t)cT * cK * 4;      // float[T*K] (combine coef per slot)
constexpr size_t OFF_XB   = 132096;                             // bf16[T*H]
constexpr size_t OFF_HB   = OFF_XB + (size_t)cT * cH * 2;       // bf16[T*K*F]
constexpr size_t OFF_PART = OFF_HB + (size_t)cT * cK * cF * 2;  // float[T*K*H]

typedef __attribute__((ext_vector_type(16))) __bf16 v16bf;
typedef __attribute__((ext_vector_type(8)))  float  v8f;

__device__ __forceinline__ v8f wmma_bf16(v16bf a, v16bf b, v8f c) {
  return __builtin_amdgcn_wmma_f32_16x16x32_bf16(false, a, false, b, (short)0, c,
                                                 false, false);
}

// A fragment (16x32 bf16): lane l (row = l&15) needs K in [abase,abase+8) and
// [16+abase,16+abase+8), abase = (l>=16)?8:0 — two contiguous 16B chunks.
__device__ __forceinline__ v16bf ldsA(const __bf16* rowp, int abase) {
  union { v16bf v; uint4 q[2]; } u;
  u.q[0] = *(const uint4*)(rowp + abase);
  u.q[1] = *(const uint4*)(rowp + 16 + abase);
  return u.v;
}
// B fragment (32x16 bf16) from [N][K]-transposed LDS tile: lane l (col = l&15)
// needs K in [kbase,kbase+8) and [16+kbase,16+kbase+8), kbase = (l>=16)?8:0.
__device__ __forceinline__ v16bf ldsB(const __bf16* p) {
  union { v16bf v; uint4 q[2]; } u;
  u.q[0] = *(const uint4*)(p);
  u.q[1] = *(const uint4*)(p + 16);
  return u.v;
}
// pack two fp32 into packed bf16 pair (lowers to v_cvt_pk_bf16_f32)
__device__ __forceinline__ unsigned pk_bf16(float lo, float hi) {
  union { __bf16 h[2]; unsigned u; } p;
  p.h[0] = (__bf16)lo; p.h[1] = (__bf16)hi;
  return p.u;
}
// CDNA5 async global->LDS copy of 32 contiguous bytes per lane (2 x B128).
// Tracked by ASYNCcnt; completes before the matching s_wait_asynccnt 0.
__device__ __forceinline__ void async_copy32(const __bf16* gsrc, __bf16* ldst) {
  unsigned dst = (unsigned)(size_t)ldst;                  // low 32 bits = LDS offset
  unsigned long long src = (unsigned long long)(size_t)gsrc;
  asm volatile("global_load_async_to_lds_b128 %0, %1, off\n\t"
               "global_load_async_to_lds_b128 %0, %1, off offset:16"
               :: "v"(dst), "v"(src) : "memory");
}
__device__ __forceinline__ void wait_async() {
  asm volatile("s_wait_asynccnt 0" ::: "memory");
}

// ---------------- 0: zero counters ----------------
__global__ void k_zero(int* __restrict__ cnt) {
  if (threadIdx.x < cE) cnt[threadIdx.x] = 0;
}

// ---------------- 1: router: softmax + top-2 ----------------
__global__ void k_router(const float* __restrict__ logits, int* __restrict__ cnt,
                         int* __restrict__ eid, float* __restrict__ tw) {
  int t = blockIdx.x * blockDim.x + threadIdx.x;
  if (t >= cT) return;
  float l[cE];
  float m = -3.0e38f;
  for (int e = 0; e < cE; ++e) { l[e] = logits[t * cE + e]; m = fmaxf(m, l[e]); }
  float s = 0.f;
  for (int e = 0; e < cE; ++e) { l[e] = __expf(l[e] - m); s += l[e]; }
  float inv = 1.f / s;
  int e0 = 0; float p0 = -1.f;
  for (int e = 0; e < cE; ++e) { float p = l[e] * inv; l[e] = p; if (p > p0) { p0 = p; e0 = e; } }
  int e1 = 0; float p1 = -1.f;
  for (int e = 0; e < cE; ++e) { if (e != e0 && l[e] > p1) { p1 = l[e]; e1 = e; } }
  eid[t * 2 + 0] = e0; tw[t * 2 + 0] = p0;
  eid[t * 2 + 1] = e1; tw[t * 2 + 1] = p1;
  atomicAdd(&cnt[e0], 1);
  atomicAdd(&cnt[e1], 1);
}

// ---------------- 2: tiny scan over E=8 ----------------
__global__ void k_scan(const int* __restrict__ cnt, int* __restrict__ bse,
                       int* __restrict__ cur) {
  if (threadIdx.x == 0) {
    int acc = 0;
    for (int e = 0; e < cE; ++e) { bse[e] = acc; cur[e] = acc; acc += cnt[e]; }
    bse[cE] = acc;
  }
}

// ---------------- 3: scatter tokens to per-expert slot lists ----------------
__global__ void k_scatter(const int* __restrict__ eid, const float* __restrict__ tw,
                          int* __restrict__ cur, int* __restrict__ stk,
                          float* __restrict__ sw) {
  int t = blockIdx.x * blockDim.x + threadIdx.x;
  if (t >= cT) return;
  for (int k = 0; k < cK; ++k) {
    int e = eid[t * 2 + k];
    int s = atomicAdd(&cur[e], 1);
    stk[s] = t * 2 + k;
    sw[s] = tw[t * 2 + k];
  }
}

// ---------------- 4: hidden_states fp32 -> bf16 ----------------
__global__ void k_cvt(const float* __restrict__ X, __bf16* __restrict__ Xb) {
  size_t i = (size_t)blockIdx.x * blockDim.x + threadIdx.x;  // T*H/8 threads
  const float4* src = (const float4*)X;
  float4 a = src[i * 2], b = src[i * 2 + 1];
  union { __bf16 h[8]; uint4 q; } o;
  o.h[0] = (__bf16)a.x; o.h[1] = (__bf16)a.y; o.h[2] = (__bf16)a.z; o.h[3] = (__bf16)a.w;
  o.h[4] = (__bf16)b.x; o.h[5] = (__bf16)b.y; o.h[6] = (__bf16)b.z; o.h[7] = (__bf16)b.w;
  ((uint4*)Xb)[i] = o.q;
}

// ---------------- 5: grouped GEMM1 (gate/up + SwiGLU), double-buffered -------
__global__ __launch_bounds__(256) void k_gemm1(
    const __bf16* __restrict__ Xb, const float* __restrict__ w1,
    const int* __restrict__ bse, const int* __restrict__ stk,
    __bf16* __restrict__ Hb) {
  constexpr int BM = 128, BN = 64, BK = 32, LDA = 40, LDB = 40;
  constexpr int ATILE = BM * LDA, BTILE = BN * LDB;
  const int e = blockIdx.z;
  const int rowStart = bse[e] + blockIdx.y * BM;
  const int slotEnd = bse[e + 1];
  if (rowStart >= slotEnd) return;
  const int n0 = blockIdx.x * BN;

  __shared__ __align__(16) __bf16 As[2 * ATILE];
  __shared__ __align__(16) __bf16 Bg[2 * BTILE];
  __shared__ __align__(16) __bf16 Bu[2 * BTILE];

  const int tid = threadIdx.x;
  const int lane = tid & 31, wave = tid >> 5;
  const int wm = wave >> 1, wn = wave & 1;          // 4x2 wave grid, 32x32 tiles
  const int lrow = lane & 15, hi = lane >> 4;
  const int abase = hi ? 8 : 0;
  const int kbase = hi ? 8 : 0;

  v8f zero = {0.f, 0.f, 0.f, 0.f, 0.f, 0.f, 0.f, 0.f};
  v8f accg[2][2], accu[2][2];
  for (int mi = 0; mi < 2; ++mi)
    for (int ni = 0; ni < 2; ++ni) { accg[mi][ni] = zero; accu[mi][ni] = zero; }

  // A gather: 2 threads/row, 32B each, async global(bf16)->LDS
  const int arow = tid >> 1, ahalf = tid & 1;
  int s = rowStart + arow; if (s >= slotEnd) s = slotEnd - 1;
  const int token = stk[s] >> 1;
  const __bf16* aSrc = Xb + (size_t)token * cH + ahalf * 16;
  const int aOff = arow * LDA + ahalf * 16;

  // B: each thread owns a 2(k) x 4(n) sub-block; pack k-pairs -> ds_store_b32
  const int bkr = (tid >> 4) * 2;       // 0,2,..,30
  const int bnc = (tid & 15) << 2;      // 0..60
  const float* Wg = w1 + ((size_t)(e * 2 + 0) * cH) * cF + n0 + bnc;
  const float* Wu = w1 + ((size_t)(e * 2 + 1) * cH) * cF + n0 + bnc;

  auto issue_tile = [&](int k0, int b) {
    async_copy32(aSrc + k0, As + b * ATILE + aOff);
    const float* g = Wg + (size_t)(k0 + bkr) * cF;
    const float* u = Wu + (size_t)(k0 + bkr) * cF;
    float4 g0 = *(const float4*)g;
    float4 g1 = *(const float4*)(g + cF);
    float4 u0 = *(const float4*)u;
    float4 u1 = *(const float4*)(u + cF);
    __bf16* dg = Bg + b * BTILE + bkr;
    __bf16* du = Bu + b * BTILE + bkr;
    *(unsigned*)(dg + (bnc + 0) * LDB) = pk_bf16(g0.x, g1.x);
    *(unsigned*)(dg + (bnc + 1) * LDB) = pk_bf16(g0.y, g1.y);
    *(unsigned*)(dg + (bnc + 2) * LDB) = pk_bf16(g0.z, g1.z);
    *(unsigned*)(dg + (bnc + 3) * LDB) = pk_bf16(g0.w, g1.w);
    *(unsigned*)(du + (bnc + 0) * LDB) = pk_bf16(u0.x, u1.x);
    *(unsigned*)(du + (bnc + 1) * LDB) = pk_bf16(u0.y, u1.y);
    *(unsigned*)(du + (bnc + 2) * LDB) = pk_bf16(u0.z, u1.z);
    *(unsigned*)(du + (bnc + 3) * LDB) = pk_bf16(u0.w, u1.w);
  };

  constexpr int nIter = cH / BK;  // 64
  issue_tile(0, 0);
  for (int it = 0; it < nIter; ++it) {
    const int cur = it & 1;
    wait_async();
    __syncthreads();
    if (it + 1 < nIter) issue_tile((it + 1) * BK, cur ^ 1);

    const __bf16* a_ = As + cur * ATILE;
    const __bf16* g_ = Bg + cur * BTILE;
    const __bf16* u_ = Bu + cur * BTILE;
    v16bf a0 = ldsA(a_ + (wm * 32 + lrow) * LDA, abase);
    v16bf a1 = ldsA(a_ + (wm * 32 + 16 + lrow) * LDA, abase);
    v16bf g0 = ldsB(g_ + (wn * 32 + lrow) * LDB + kbase);
    v16bf g1 = ldsB(g_ + (wn * 32 + 16 + lrow) * LDB + kbase);
    v16bf u0 = ldsB(u_ + (wn * 32 + lrow) * LDB + kbase);
    v16bf u1 = ldsB(u_ + (wn * 32 + 16 + lrow) * LDB + kbase);

    accg[0][0] = wmma_bf16(a0, g0, accg[0][0]);
    accg[0][1] = wmma_bf16(a0, g1, accg[0][1]);
    accg[1][0] = wmma_bf16(a1, g0, accg[1][0]);
    accg[1][1] = wmma_bf16(a1, g1, accg[1][1]);
    accu[0][0] = wmma_bf16(a0, u0, accu[0][0]);
    accu[0][1] = wmma_bf16(a0, u1, accu[0][1]);
    accu[1][0] = wmma_bf16(a1, u0, accu[1][0]);
    accu[1][1] = wmma_bf16(a1, u1, accu[1][1]);
  }

  // epilogue: SwiGLU, store bf16 h by slot row
  for (int mi = 0; mi < 2; ++mi)
    for (int ni = 0; ni < 2; ++ni) {
      v8f g = accg[mi][ni], u = accu[mi][ni];
      for (int v = 0; v < 8; ++v) {
        int M = wm * 32 + mi * 16 + v + (hi ? 8 : 0);
        int srow = rowStart + M;
        if (srow < slotEnd) {
          int N = n0 + wn * 32 + ni * 16 + lrow;
          float gv = g[v], uv = u[v];
          float h = gv * uv / (1.f + __expf(-gv));
          Hb[(size_t)srow * cF + N] = (__bf16)h;
        }
      }
    }
}

// ---------------- 6: grouped GEMM2 (down proj, scaled), double-buffered ------
__global__ __launch_bounds__(256) void k_gemm2(
    const __bf16* __restrict__ Hb, const float* __restrict__ w2,
    const int* __restrict__ bse, const int* __restrict__ stk,
    const float* __restrict__ sw, float* __restrict__ part) {
  constexpr int BM = 128, BN = 128, BK = 32, LDA = 40, LDB = 40;
  constexpr int ATILE = BM * LDA, BTILE = BN * LDB;
  const int e = blockIdx.z;
  const int rowStart = bse[e] + blockIdx.y * BM;
  const int slotEnd = bse[e + 1];
  if (rowStart >= slotEnd) return;
  const int n0 = blockIdx.x * BN;

  __shared__ __align__(16) __bf16 As[2 * ATILE];
  __shared__ __align__(16) __bf16 Bs[2 * BTILE];

  const int tid = threadIdx.x;
  const int lane = tid & 31, wave = tid >> 5;
  const int wm = wave >> 1, wn = wave & 1;          // 4x2 wave grid, 32x64 tiles
  const int lrow = lane & 15, hi = lane >> 4;
  const int abase = hi ? 8 : 0;
  const int kbase = hi ? 8 : 0;

  v8f zero = {0.f, 0.f, 0.f, 0.f, 0.f, 0.f, 0.f, 0.f};
  v8f acc[2][4];
  for (int mi = 0; mi < 2; ++mi)
    for (int ni = 0; ni < 4; ++ni) acc[mi][ni] = zero;

  const int arow = tid >> 1, ahalf = tid & 1;
  int s = rowStart + arow; if (s >= slotEnd) s = slotEnd - 1;
  const __bf16* aSrc = Hb + (size_t)s * cF + ahalf * 16;
  const int aOff = arow * LDA + ahalf * 16;

  const float* W = w2 + ((size_t)e * cF) * cH + n0;

  auto issue_tile = [&](int k0, int b) {
    async_copy32(aSrc + k0, As + b * ATILE + aOff);
    // 32k x 128n tile: 2 sub-blocks of 2(k) x 4(n) per thread
    for (int i = 0; i < 2; ++i) {
      int idx = tid + i * 256;
      int kr = (idx >> 5) * 2;            // 0,2,..,30
      int nc = (idx & 31) << 2;           // 0..124
      const float* p = W + (size_t)(k0 + kr) * cH + nc;
      float4 f0 = *(const float4*)p;
      float4 f1 = *(const float4*)(p + cH);
      __bf16* d = Bs + b * BTILE + kr;
      *(unsigned*)(d + (nc + 0) * LDB) = pk_bf16(f0.x, f1.x);
      *(unsigned*)(d + (nc + 1) * LDB) = pk_bf16(f0.y, f1.y);
      *(unsigned*)(d + (nc + 2) * LDB) = pk_bf16(f0.z, f1.z);
      *(unsigned*)(d + (nc + 3) * LDB) = pk_bf16(f0.w, f1.w);
    }
  };

  constexpr int nIter = cF / BK;  // 44
  issue_tile(0, 0);
  for (int it = 0; it < nIter; ++it) {
    const int cur = it & 1;
    wait_async();
    __syncthreads();
    if (it + 1 < nIter) issue_tile((it + 1) * BK, cur ^ 1);

    const __bf16* a_ = As + cur * ATILE;
    const __bf16* b_ = Bs + cur * BTILE;
    v16bf a0 = ldsA(a_ + (wm * 32 + lrow) * LDA, abase);
    v16bf a1 = ldsA(a_ + (wm * 32 + 16 + lrow) * LDA, abase);
    v16bf b0 = ldsB(b_ + (wn * 64 + lrow) * LDB + kbase);
    v16bf b1 = ldsB(b_ + (wn * 64 + 16 + lrow) * LDB + kbase);
    v16bf b2 = ldsB(b_ + (wn * 64 + 32 + lrow) * LDB + kbase);
    v16bf b3 = ldsB(b_ + (wn * 64 + 48 + lrow) * LDB + kbase);

    acc[0][0] = wmma_bf16(a0, b0, acc[0][0]);
    acc[0][1] = wmma_bf16(a0, b1, acc[0][1]);
    acc[0][2] = wmma_bf16(a0, b2, acc[0][2]);
    acc[0][3] = wmma_bf16(a0, b3, acc[0][3]);
    acc[1][0] = wmma_bf16(a1, b0, acc[1][0]);
    acc[1][1] = wmma_bf16(a1, b1, acc[1][1]);
    acc[1][2] = wmma_bf16(a1, b2, acc[1][2]);
    acc[1][3] = wmma_bf16(a1, b3, acc[1][3]);
  }

  // epilogue: scale by combine coef, scatter to partial[(t*2+k), H]
  for (int mi = 0; mi < 2; ++mi)
    for (int ni = 0; ni < 4; ++ni) {
      v8f c = acc[mi][ni];
      for (int v = 0; v < 8; ++v) {
        int M = wm * 32 + mi * 16 + v + (hi ? 8 : 0);
        int srow = rowStart + M;
        if (srow < slotEnd) {
          int N = n0 + wn * 64 + ni * 16 + lrow;
          int tk = stk[srow];
          float coef = sw[srow];
          part[(size_t)tk * cH + N] = coef * c[v];
        }
      }
    }
}

// ---------------- 7: reduce K=2 partials ----------------
__global__ void k_reduce(const float* __restrict__ part, float* __restrict__ out) {
  size_t i = (size_t)blockIdx.x * blockDim.x + threadIdx.x;  // T*H/4 threads
  constexpr int RL = cH / 4;
  size_t t = i / RL, c = i % RL;
  const float4* p = (const float4*)part;
  float4 a = p[(2 * t) * RL + c];
  float4 b = p[(2 * t + 1) * RL + c];
  float4 o; o.x = a.x + b.x; o.y = a.y + b.y; o.z = a.z + b.z; o.w = a.w + b.w;
  ((float4*)out)[i] = o;
}

extern "C" void kernel_launch(void* const* d_in, const int* in_sizes, int n_in,
                              void* d_out, int out_size, void* d_ws, size_t ws_size,
                              hipStream_t stream) {
  const float* X      = (const float*)d_in[0];
  const float* logits = (const float*)d_in[1];
  const float* w1     = (const float*)d_in[2];
  const float* w2     = (const float*)d_in[3];
  float* out = (float*)d_out;

  char* w = (char*)d_ws;
  int*    cnt  = (int*)(w + OFF_CNT);
  int*    bse  = (int*)(w + OFF_BASE);
  int*    cur  = (int*)(w + OFF_CUR);
  int*    eid  = (int*)(w + OFF_EID);
  float*  tw   = (float*)(w + OFF_TW);
  int*    stk  = (int*)(w + OFF_STK);
  float*  sw   = (float*)(w + OFF_SW);
  __bf16* Xb   = (__bf16*)(w + OFF_XB);
  __bf16* Hb   = (__bf16*)(w + OFF_HB);
  float*  part = (float*)(w + OFF_PART);

  k_zero<<<1, 32, 0, stream>>>(cnt);
  k_router<<<cT / 256, 256, 0, stream>>>(logits, cnt, eid, tw);
  k_scan<<<1, 1, 0, stream>>>(cnt, bse, cur);
  k_scatter<<<cT / 256, 256, 0, stream>>>(eid, tw, cur, stk, sw);
  k_cvt<<<(cT * cH / 8) / 256, 256, 0, stream>>>(X, Xb);
  k_gemm1<<<dim3(cF / 64, cT / 128, cE), 256, 0, stream>>>(Xb, w1, bse, stk, Hb);
  k_gemm2<<<dim3(cH / 128, cT / 128, cE), 256, 0, stream>>>(Hb, w2, bse, stk, sw, part);
  k_reduce<<<(cT * cH / 4) / 256, 256, 0, stream>>>(part, out);
}
